// Model_33483565039866
// MI455X (gfx1250) — matrix operand order repaired and checked by reading, not compiled, not run on hardware
//
#include <hip/hip_runtime.h>
#include <hip/hip_bf16.h>

typedef __attribute__((ext_vector_type(16))) _Float16 v16h;
typedef __attribute__((ext_vector_type(8)))  float    v8f;

// ---------------- dims ----------------
#define BB 8
#define SEQ 512
#define NV 21
#define PRED 96
#define DM 128
#define DSZ 16     // d_state
#define DI 256     // d_inner
#define DTR 8
#define PNUM 64
#define PLEN 16
#define MROWS (BB*NV)          // 168
#define MTOK  (BB*NV*PNUM)     // 10752
#define LVAR  (NV*PNUM)        // 1344

// ================= WMMA fragment helpers =================
__device__ __forceinline__ v16h pack16(float4 a, float4 b, float4 c, float4 d) {
  v16h f;
  f[0]=(_Float16)a.x; f[1]=(_Float16)a.y; f[2]=(_Float16)a.z; f[3]=(_Float16)a.w;
  f[4]=(_Float16)b.x; f[5]=(_Float16)b.y; f[6]=(_Float16)b.z; f[7]=(_Float16)b.w;
  f[8]=(_Float16)c.x; f[9]=(_Float16)c.y; f[10]=(_Float16)c.z; f[11]=(_Float16)c.w;
  f[12]=(_Float16)d.x; f[13]=(_Float16)d.y; f[14]=(_Float16)d.z; f[15]=(_Float16)d.w;
  return f;
}
// B fragment fast path: p = rowptr + kloB (kloB = (lane>=16)?8:0)
__device__ __forceinline__ v16h frag_b_fast(const float* __restrict__ p, int k0) {
  const float4* q = (const float4*)(p + k0);
  return pack16(q[0], q[1], q[4], q[5]);     // K 0..7, 16..23 relative to klo
}
// tail paths with zero fill
__device__ __forceinline__ v16h frag_a_tail(const float* __restrict__ base, int K, int k0, int lane) {
  v16h f;
  int klo = (lane >> 4) << 3;
#pragma unroll
  for (int j = 0; j < 16; ++j) {
    int k = k0 + klo + (j < 8 ? j : 8 + j);
    f[j] = (k < K) ? (_Float16)base[k] : (_Float16)0.f;
  }
  return f;
}
__device__ __forceinline__ v16h frag_b_tail(const float* __restrict__ base, int K, int k0, int lane) {
  v16h f;
  int klo = (lane >> 4) << 3;
#pragma unroll
  for (int j = 0; j < 16; ++j) {
    int k = k0 + klo + (j < 8 ? j : 8 + j);
    f[j] = (k < K) ? (_Float16)base[k] : (_Float16)0.f;
  }
  return f;
}

// gfx1250 async global->LDS copy (ASYNCcnt-tracked)
__device__ __forceinline__ void async_ld_b128(unsigned ldsoff, const float* g) {
  asm volatile("global_load_async_to_lds_b128 %0, %1, off" :: "v"(ldsoff), "v"(g) : "memory");
}

// ================= generic WMMA GEMM =================
// C[m,n] = act( sum_k A[rowmap(m)*lda + k] * W[n*K + k] + bias[n] )
// rowmap(m) = (m/rgroup)*(rgroup+rpad) + m%rgroup   (padded im2col conv)
// Block = 8 waves; all waves share one 32-row A tile, staged to LDS via async
// copies (double buffered); each wave owns one 16-col B tile and computes a
// 32x16 output tile (two WMMA accumulator chains sharing the B fragment).
// act: 0 none, 1 relu, 3 softplus
__global__ void wmma_gemm_kernel(const float* __restrict__ A, int lda, int rgroup, int rpad,
                                 const float* __restrict__ W,
                                 const float* __restrict__ bias,
                                 float* __restrict__ C, int ldc,
                                 int M, int Nout, int K, int act) {
  __shared__ float tA[2][32*32];          // 2 x 4KB double buffer
  int tid  = threadIdx.x;
  int wave = tid >> 5;
  int lane = tid & 31;
  int tm = blockIdx.x;                    // rows [tm*32, tm*32+32)
  int tn = blockIdx.y * 8 + wave;
  int r  = lane & 15;
  int kloA = (lane >> 4) << 3;
  int kloB = (lane >> 4) << 3;

  // staging source: thread t copies 16B chunk (row = t/8, chunk = t%8) of the tile
  int sm = tm * 32 + (tid >> 3);
  int sam = sm < M ? sm : M - 1;
  int sar = (sam / rgroup) * (rgroup + rpad) + (sam % rgroup);
  const float* gsrc = A + (long)sar * lda + ((tid & 7) << 2);
  unsigned lds0 = (unsigned)(unsigned long long)(&tA[0][0]) + (unsigned)tid * 16u;
  unsigned lds1 = (unsigned)(unsigned long long)(&tA[1][0]) + (unsigned)tid * 16u;

  // A-fragment row mapping (used by the tail loads)
  int m0 = tm * 32 + r;
  int m1 = m0 + 16;
  int am0 = m0 < M ? m0 : M - 1;
  int am1 = m1 < M ? m1 : M - 1;
  int ar0 = (am0 / rgroup) * (rgroup + rpad) + (am0 % rgroup);
  int ar1 = (am1 / rgroup) * (rgroup + rpad) + (am1 % rgroup);

  int nrow = tn * 16 + r;
  int wn = nrow < Nout ? nrow : Nout - 1;
  const float* Bp = W + (long)wn * K + kloB;

  v8f acc0 = {0.f,0.f,0.f,0.f,0.f,0.f,0.f,0.f};
  v8f acc1 = {0.f,0.f,0.f,0.f,0.f,0.f,0.f,0.f};

  int kmain = K & ~31;
  int nk = kmain >> 5;
  if (nk > 0) async_ld_b128(lds0, gsrc);                 // stage k-step 0
  for (int i = 0; i < nk; ++i) {
    int cur = i & 1;
    if (i + 1 < nk) {                                    // stage next step
      async_ld_b128(cur ? lds0 : lds1, gsrc + ((i + 1) << 5));
      asm volatile("s_wait_asynccnt 0x1" ::: "memory");  // oldest stage done
    } else {
      asm volatile("s_wait_asynccnt 0x0" ::: "memory");
    }
    __syncthreads();                                     // all waves' stage visible
    int k0 = i << 5;
    if (k0 + 160 <= kmain)
      __builtin_prefetch((const void*)(Bp + k0 + 128), 0, 1);
    v16h bf  = frag_b_fast(Bp, k0);
    const float* tp = &tA[cur][0];
    const float4* qa = (const float4*)(tp + r * 32 + kloA);
    v16h af0 = pack16(qa[0], qa[1], qa[4], qa[5]);
    const float4* qb = (const float4*)(tp + (r + 16) * 32 + kloA);
    v16h af1 = pack16(qb[0], qb[1], qb[4], qb[5]);
    acc0 = __builtin_amdgcn_wmma_f32_16x16x32_f16(false, af0, false, bf, (short)0, acc0, false, false);
    acc1 = __builtin_amdgcn_wmma_f32_16x16x32_f16(false, af1, false, bf, (short)0, acc1, false, false);
    __syncthreads();                                     // protect buffer reuse
  }
  if (kmain < K) {                                       // zero-filled tail step
    const float* Ab0 = A + (long)ar0 * lda;
    const float* Ab1 = A + (long)ar1 * lda;
    const float* Bb  = W + (long)wn * K;
    v16h bf  = frag_b_tail(Bb, K, kmain, lane);
    v16h af0 = frag_a_tail(Ab0, K, kmain, lane);
    v16h af1 = frag_a_tail(Ab1, K, kmain, lane);
    acc0 = __builtin_amdgcn_wmma_f32_16x16x32_f16(false, af0, false, bf, (short)0, acc0, false, false);
    acc1 = __builtin_amdgcn_wmma_f32_16x16x32_f16(false, af1, false, bf, (short)0, acc1, false, false);
  }

  int crow0 = tm * 32 + ((lane >> 4) << 3);
  int ccol  = tn * 16 + (lane & 15);
  if (ccol < Nout) {
    float bv = bias ? bias[ccol] : 0.f;
#pragma unroll
    for (int v = 0; v < 8; ++v) {
      int ra = crow0 + v;
      if (ra < M) {
        float x = acc0[v] + bv;
        if (act == 1) x = fmaxf(x, 0.f);
        else if (act == 3) x = (x > 20.f) ? x : __logf(1.f + __expf(x));
        C[(long)ra * ldc + ccol] = x;
      }
      int rb2 = ra + 16;
      if (rb2 < M) {
        float x = acc1[v] + bv;
        if (act == 1) x = fmaxf(x, 0.f);
        else if (act == 3) x = (x > 20.f) ? x : __logf(1.f + __expf(x));
        C[(long)rb2 * ldc + ccol] = x;
      }
    }
  }
}

// ================= RevIN normalize + transpose to [B,N,L] =================
__global__ void revin_kernel(const float* __restrict__ x, const float* __restrict__ rw,
                             const float* __restrict__ rb, float* __restrict__ xt,
                             float* __restrict__ meanp, float* __restrict__ stdp) {
  int bn = blockIdx.x;               // [0,168)
  int b = bn / NV, n = bn % NV;
  __shared__ float red[256];
  __shared__ float sm2[2];
  float s = 0.f, s2 = 0.f;
  for (int l = threadIdx.x; l < SEQ; l += 256) {
    float v = x[((long)(b*SEQ + l))*NV + n];
    s += v; s2 += v*v;
  }
  red[threadIdx.x] = s; __syncthreads();
  for (int o = 128; o > 0; o >>= 1) { if (threadIdx.x < o) red[threadIdx.x] += red[threadIdx.x+o]; __syncthreads(); }
  if (threadIdx.x == 0) sm2[0] = red[0] / (float)SEQ;
  __syncthreads();
  red[threadIdx.x] = s2; __syncthreads();
  for (int o = 128; o > 0; o >>= 1) { if (threadIdx.x < o) red[threadIdx.x] += red[threadIdx.x+o]; __syncthreads(); }
  if (threadIdx.x == 0) sm2[1] = red[0] / (float)SEQ;
  __syncthreads();
  float mean = sm2[0];
  float var  = sm2[1] - mean*mean;
  float sd = sqrtf(var + 1e-5f);
  if (threadIdx.x == 0) { meanp[bn] = mean; stdp[bn] = sd; }
  float g = rw[n], bb = rb[n];
  for (int l = threadIdx.x; l < SEQ; l += 256) {
    float v = x[((long)(b*SEQ + l))*NV + n];
    xt[(long)bn*SEQ + l] = (v - mean) / sd * g + bb;
  }
}

// ================= LayerNorm over D=128 (optional residual add) =================
__global__ void ln_kernel(const float* __restrict__ x, const float* __restrict__ res,
                          const float* __restrict__ g, const float* __restrict__ bta,
                          float* __restrict__ out) {
  long row = blockIdx.x;
  int t = threadIdx.x;               // 128 threads
  float v = x[row*DM + t] + (res ? res[row*DM + t] : 0.f);
  __shared__ float red[DM];
  red[t] = v; __syncthreads();
  for (int o = 64; o > 0; o >>= 1) { if (t < o) red[t] += red[t+o]; __syncthreads(); }
  float mean = red[0] / (float)DM;
  __syncthreads();
  float c = v - mean;
  red[t] = c*c; __syncthreads();
  for (int o = 64; o > 0; o >>= 1) { if (t < o) red[t] += red[t+o]; __syncthreads(); }
  float var = red[0] / (float)DM;
  out[row*DM + t] = c * rsqrtf(var + 1e-5f) * g[t] + bta[t];
}

// ================= mamba pieces =================
__global__ void dwconv_silu_kernel(const float* __restrict__ xz, int ldi,
                                   const float* __restrict__ cw, const float* __restrict__ cb,
                                   float* __restrict__ xo, int Bn, int L) {
  long idx = (long)blockIdx.x*256 + threadIdx.x;
  long tot = (long)Bn*L*DI;
  if (idx >= tot) return;
  int d = idx % DI; long bl = idx / DI; int l = bl % L; int b = bl / L;
  float acc = cb[d];
#pragma unroll
  for (int j = 0; j < 4; ++j) {
    int l2 = l - 3 + j;
    if (l2 >= 0) acc += xz[((long)(b*L + l2))*ldi + d] * cw[d*4 + j];
  }
  xo[idx] = acc / (1.f + __expf(-acc));
}

__global__ void scan_kernel(const float* __restrict__ delta, const float* __restrict__ xi,
                            const float* __restrict__ dbc, const float* __restrict__ A_log,
                            const float* __restrict__ Dp, float* __restrict__ y, int L) {
  int b = blockIdx.x;
  int d = threadIdx.x;               // 256
  float A[DSZ], h[DSZ];
#pragma unroll
  for (int s = 0; s < DSZ; ++s) { A[s] = -__expf(A_log[d*DSZ + s]); h[s] = 0.f; }
  float Dd = Dp[d];
  __shared__ float sB[DSZ], sC[DSZ];
  for (int l = 0; l < L; ++l) {
    long row = (long)b*L + l;
    if (d < DSZ) sB[d] = dbc[row*40 + DTR + d];
    else if (d < 2*DSZ) sC[d-DSZ] = dbc[row*40 + DTR + DSZ + (d-DSZ)];
    __syncthreads();
    float dl = delta[row*DI + d];
    float xv = xi[row*DI + d];
    float dx = dl * xv;
    float acc = 0.f;
#pragma unroll
    for (int s = 0; s < DSZ; ++s) {
      h[s] = h[s]*__expf(dl*A[s]) + dx*sB[s];
      acc += h[s]*sC[s];
    }
    y[row*DI + d] = acc + xv*Dd;
    __syncthreads();
  }
}

__global__ void silu_mul_kernel(const float* __restrict__ y, const float* __restrict__ z, int ldz,
                                float* __restrict__ u, long n) {
  long idx = (long)blockIdx.x*256 + threadIdx.x;
  if (idx >= n) return;
  int d = idx % DI; long m = idx / DI;
  float zv = z[m*ldz + d];
  u[idx] = y[idx] * zv / (1.f + __expf(-zv));
}

// ================= misc elementwise =================
__global__ void add_kernel(const float* __restrict__ a, const float* __restrict__ b,
                           float* __restrict__ o, long n) {
  long idx = (long)blockIdx.x*256 + threadIdx.x;
  if (idx < n) o[idx] = a[idx] + b[idx];
}

__global__ void pos_kernel(float* __restrict__ pe) {
  int idx = blockIdx.x*256 + threadIdx.x;
  if (idx >= PNUM*DM) return;
  int p = idx / DM, d = idx % DM;
  int i = d >> 1;
  float div = __expf(-(float)(2*i) * __logf(10000.f) / (float)DM);
  float a = (float)p * div;
  pe[idx] = (d & 1) ? __cosf(a) : __sinf(a);
}

__global__ void addpos_kernel(float* __restrict__ enc, const float* __restrict__ pe) {
  long idx = (long)blockIdx.x*256 + threadIdx.x;
  if (idx >= (long)MTOK*DM) return;
  int d = idx % DM; long m = idx / DM; int p = m % PNUM;
  enc[idx] += pe[p*DM + d];
}

__global__ void patch_kernel(const float* __restrict__ xt, float* __restrict__ patches) {
  long idx = (long)blockIdx.x*256 + threadIdx.x;
  if (idx >= (long)MTOK*PLEN) return;
  int j = idx % PLEN; long m = idx / PLEN;
  int p = m % PNUM; long bn = m / PNUM;
  int l = p*8 + j; if (l > SEQ-1) l = SEQ-1;   // ReplicationPad1d
  patches[idx] = xt[bn*SEQ + l];
}

__global__ void pad_kernel(const float* __restrict__ enc, float* __restrict__ encpad) {
  long idx = (long)blockIdx.x*256 + threadIdx.x;
  long tot = (long)MROWS*(PNUM+2)*DM;
  if (idx >= tot) return;
  int d = idx % DM; long rp = idx / DM; int p = rp % (PNUM+2); long bn = rp / (PNUM+2);
  encpad[idx] = (p == 0 || p == PNUM+1) ? 0.f : enc[(bn*PNUM + (p-1))*DM + d];
}

__global__ void convw_repack_kernel(const float* __restrict__ w, float* __restrict__ w2) {
  int idx = blockIdx.x*256 + threadIdx.x;
  if (idx >= DM*3*DM) return;
  int o = idx / (3*DM); int k = idx % (3*DM);
  int t = k / DM; int di = k % DM;
  w2[idx] = w[(o*DM + di)*3 + t];
}

__global__ void headw_repack_kernel(const float* __restrict__ w, float* __restrict__ w2) {
  long idx = (long)blockIdx.x*256 + threadIdx.x;
  if (idx >= (long)PRED*DM*PNUM) return;
  int o = idx / (DM*PNUM); int k = idx % (DM*PNUM);
  int p = k / DM; int d = k % DM;
  w2[idx] = w[(long)o*DM*PNUM + d*PNUM + p];
}

// reference reshape [B,N,D,P] -> [B, P*N, D]: token(n,j), e -> enc_t[b,n,p=e%64,d=2j+e/64]
__global__ void gather_vin_kernel(const float* __restrict__ enc_t, float* __restrict__ vin) {
  long idx = (long)blockIdx.x*256 + threadIdx.x;
  if (idx >= (long)BB*LVAR*DM) return;
  int e = idx % DM; long t = idx / DM;
  int tok = t % LVAR; int b = t / LVAR;
  int n = tok / PNUM; int j = tok % PNUM;
  int p = e & 63; int dch = 2*j + (e >> 6);
  vin[idx] = enc_t[(((long)(b*NV + n))*PNUM + p)*DM + dch];
}

__global__ void final_out_kernel(const float* __restrict__ xv, const float* __restrict__ xp,
                                 const float* __restrict__ rw, const float* __restrict__ rb,
                                 const float* __restrict__ meanp, const float* __restrict__ stdp,
                                 float* __restrict__ out) {
  int idx = blockIdx.x*256 + threadIdx.x;
  if (idx >= BB*PRED*NV) return;
  int n = idx % NV; int bt = idx / NV; int t = bt % PRED; int b = bt / PRED;
  int bn = b*NV + n;
  float v = xv[bn*PRED + t] + xp[bn*PRED + t];
  v = (v - rb[n]) / (rw[n] + 1e-10f);
  out[idx] = v * stdp[bn] + meanp[bn];
}

// ================= host orchestration =================
struct MambaP {
  const float *in_w,*in_b,*conv_w,*conv_b,*xproj_w,*dt_w,*dt_b,*A_log,*D,*out_w,*out_b,
              *ln1_g,*ln1_b,*ln2_g,*ln2_b,*f1w,*f1b,*f2w,*f2b;
};

extern "C" void kernel_launch(void* const* d_in, const int* in_sizes, int n_in,
                              void* d_out, int out_size, void* d_ws, size_t ws_size,
                              hipStream_t stream) {
  (void)in_sizes; (void)n_in; (void)out_size; (void)ws_size;

  // ---- inputs / params (setup_inputs dict insertion order) ----
  const float* x_enc = (const float*)d_in[0];
  int pi = 4;
  auto nxt = [&]() { return (const float*)d_in[pi++]; };
  auto readM = [&]() {
    MambaP m;
    m.in_w=nxt(); m.in_b=nxt(); m.conv_w=nxt(); m.conv_b=nxt(); m.xproj_w=nxt();
    m.dt_w=nxt(); m.dt_b=nxt(); m.A_log=nxt(); m.D=nxt(); m.out_w=nxt(); m.out_b=nxt();
    m.ln1_g=nxt(); m.ln1_b=nxt(); m.ln2_g=nxt(); m.ln2_b=nxt();
    m.f1w=nxt(); m.f1b=nxt(); m.f2w=nxt(); m.f2b=nxt();
    return m;
  };
  const float* revin_w = nxt();
  const float* revin_b = nxt();
  const float* emb_w = nxt();
  const float* emb_b = nxt();
  MambaP m1[2]; m1[0] = readM(); m1[1] = readM();
  const float* proj_w = nxt();
  const float* proj_b = nxt();
  const float* val_w  = nxt();
  const float *tem_cw[2], *tem_cb[2], *tem_lg[2], *tem_lb[2];
  for (int i = 0; i < 2; ++i) { tem_cw[i]=nxt(); tem_cb[i]=nxt(); tem_lg[i]=nxt(); tem_lb[i]=nxt(); }
  MambaP vm[2]; const float *var_lg[2], *var_lb[2];
  for (int i = 0; i < 2; ++i) { vm[i] = readM(); var_lg[i]=nxt(); var_lb[i]=nxt(); }
  const float* ln2_g = nxt(); const float* ln2_b = nxt();
  const float* ln3_g = nxt(); const float* ln3_b = nxt();
  const float* head_w = nxt(); const float* head_b = nxt();

  // ---- workspace carve ----
  float* base = (float*)d_ws;
  size_t off = 0;
  auto alloc = [&](size_t n) { float* p = base + off; off += (n + 15) & ~(size_t)15; return p; };
  float* XT     = alloc((size_t)MROWS*SEQ);
  float* MEAN   = alloc(MROWS);
  float* STD    = alloc(MROWS);
  float* EMB    = alloc((size_t)MROWS*DM);
  float* XV     = alloc((size_t)MROWS*PRED);
  float* XPATCH = alloc((size_t)MROWS*PRED);
  float* PE     = alloc((size_t)PNUM*DM);
  float* ENC    = alloc((size_t)MTOK*DM);
  float* ENCPAD = alloc((size_t)MROWS*(PNUM+2)*DM);
  float* CBUF   = alloc((size_t)MTOK*DM);
  float* TBUF   = alloc((size_t)MTOK*DM);
  float* VIN    = alloc((size_t)MTOK*DM);
  float* CONVW2 = alloc((size_t)DM*3*DM);
  float* HEADW2 = alloc((size_t)PRED*DM*PNUM);
  float* PATCH  = alloc((size_t)MTOK*PLEN);
  float* G_XZ   = alloc((size_t)MTOK*2*DI);
  float* G_XI   = alloc((size_t)MTOK*DI);
  float* G_DBC  = alloc((size_t)MTOK*40);
  float* G_DLT  = alloc((size_t)MTOK*DI);
  float* G_Y    = alloc((size_t)MTOK*DI);
  float* G_U    = alloc((size_t)MTOK*DI);
  float* G_MO   = alloc((size_t)MTOK*DM);
  float* G_X1   = alloc((size_t)MTOK*DM);
  float* G_FFH  = alloc((size_t)MTOK*DI);   // dff <= 256
  float* G_FFO  = alloc((size_t)MTOK*DM);

  auto gemm = [&](const float* A, int lda, int rg, int rp, const float* W, const float* bias,
                  float* C, int ldc, int M, int N, int K, int act) {
    dim3 grid((M+31)/32, ((N+15)/16 + 7)/8);
    wmma_gemm_kernel<<<grid, 256, 0, stream>>>(A, lda, rg, rp, W, bias, C, ldc, M, N, K, act);
  };
  auto blocks = [](long n) { return (unsigned)((n + 255) / 256); };

  auto mamba_ffn = [&](const MambaP& P, float* x, float* out, int Bn, int L, int dff) {
    int M = Bn * L;
    gemm(x, DM, M, 0, P.in_w, P.in_b, G_XZ, 2*DI, M, 2*DI, DM, 0);
    dwconv_silu_kernel<<<blocks((long)M*DI), 256, 0, stream>>>(G_XZ, 2*DI, P.conv_w, P.conv_b, G_XI, Bn, L);
    gemm(G_XI, DI, M, 0, P.xproj_w, nullptr, G_DBC, 40, M, 40, DI, 0);
    gemm(G_DBC, 40, M, 0, P.dt_w, P.dt_b, G_DLT, DI, M, DI, DTR, 3);  // softplus
    scan_kernel<<<Bn, DI, 0, stream>>>(G_DLT, G_XI, G_DBC, P.A_log, P.D, G_Y, L);
    silu_mul_kernel<<<blocks((long)M*DI), 256, 0, stream>>>(G_Y, G_XZ + DI, 2*DI, G_U, (long)M*DI);
    gemm(G_U, DI, M, 0, P.out_w, P.out_b, G_MO, DM, M, DM, DI, 0);
    ln_kernel<<<M, DM, 0, stream>>>(x, G_MO, P.ln1_g, P.ln1_b, G_X1);
    gemm(G_X1, DM, M, 0, P.f1w, P.f1b, G_FFH, dff, M, dff, DM, 1); // relu
    gemm(G_FFH, dff, M, 0, P.f2w, P.f2b, G_FFO, DM, M, DM, dff, 0);
    ln_kernel<<<M, DM, 0, stream>>>(G_X1, G_FFO, P.ln2_g, P.ln2_b, out);
  };

  // ---- 1. RevIN + transpose to [B,N,L] ----
  revin_kernel<<<MROWS, 256, 0, stream>>>(x_enc, revin_w, revin_b, XT, MEAN, STD);

  // ---- 2. inverted embedding + mamba1 stack + projection ----
  gemm(XT, SEQ, MROWS, 0, emb_w, emb_b, EMB, DM, MROWS, DM, SEQ, 0);
  for (int i = 0; i < 2; ++i) mamba_ffn(m1[i], EMB, EMB, BB, NV, 16);
  gemm(EMB, DM, MROWS, 0, proj_w, proj_b, XV, PRED, MROWS, PRED, DM, 0);

  // ---- 3. patch embedding: unfold -> GEMM -> +pos ----
  patch_kernel<<<blocks((long)MTOK*PLEN), 256, 0, stream>>>(XT, PATCH);
  gemm(PATCH, PLEN, MTOK, 0, val_w, nullptr, ENC, DM, MTOK, DM, PLEN, 0);
  pos_kernel<<<blocks(PNUM*DM), 256, 0, stream>>>(PE);
  addpos_kernel<<<blocks((long)MTOK*DM), 256, 0, stream>>>(ENC, PE);

  // ---- 4. encoder layers: TemPatch conv (im2col WMMA) + VarPatch mamba ----
  for (int i = 0; i < 2; ++i) {
    pad_kernel<<<blocks((long)MROWS*(PNUM+2)*DM), 256, 0, stream>>>(ENC, ENCPAD);
    convw_repack_kernel<<<blocks(DM*3*DM), 256, 0, stream>>>(tem_cw[i], CONVW2);
    gemm(ENCPAD, DM, PNUM, 2, CONVW2, tem_cb[i], CBUF, DM, MTOK, DM, 3*DM, 0);
    ln_kernel<<<MTOK, DM, 0, stream>>>(CBUF, nullptr, tem_lg[i], tem_lb[i], CBUF);
    add_kernel<<<blocks((long)MTOK*DM), 256, 0, stream>>>(CBUF, ENC, TBUF, (long)MTOK*DM);  // enc_t
    gather_vin_kernel<<<blocks((long)MTOK*DM), 256, 0, stream>>>(TBUF, VIN);
    mamba_ffn(vm[i], VIN, VIN, BB, LVAR, DI);
    ln_kernel<<<MTOK, DM, 0, stream>>>(VIN, nullptr, var_lg[i], var_lb[i], VIN);
    add_kernel<<<blocks((long)MTOK*DM), 256, 0, stream>>>(VIN, TBUF, CBUF, (long)MTOK*DM);
    ln_kernel<<<MTOK, DM, 0, stream>>>(CBUF, nullptr, i == 0 ? ln2_g : ln3_g,
                                       i == 0 ? ln2_b : ln3_b, ENC);
  }

  // ---- 5. head + combine + RevIN denorm ----
  headw_repack_kernel<<<blocks((long)PRED*DM*PNUM), 256, 0, stream>>>(head_w, HEADW2);
  gemm(ENC, DM*PNUM, MROWS, 0, HEADW2, head_b, XPATCH, PRED, MROWS, PRED, DM*PNUM, 0);
  final_out_kernel<<<blocks(BB*PRED*NV), 256, 0, stream>>>(XV, XPATCH, revin_w, revin_b,
                                                           MEAN, STD, (float*)d_out);
}
